// Gaug_3504693313818
// MI455X (gfx1250) — matrix operand order repaired and checked by reading, not compiled, not run on hardware
//
#include <hip/hip_runtime.h>
#include <hip/hip_bf16.h>
#include <math.h>

// Problem constants (match reference)
#define NNODE 4096
#define FIN   256
#define HID   128
#define NCLS  32
#define NEDGE 131072
#define NWRD  (NNODE / 32)   // mask words per row
#define EPSF  1e-6f

typedef __attribute__((ext_vector_type(16))) __bf16 v16bf;
typedef __attribute__((ext_vector_type(8)))  float  v8f;

// ---------- helpers ----------
__device__ __forceinline__ float sigmoidf_(float x) { return 1.f / (1.f + expf(-x)); }

// monotone float -> uint encoding for atomicMax over signed floats
__device__ __forceinline__ unsigned fenc(float x) {
  unsigned b = __float_as_uint(x);
  return (b & 0x80000000u) ? ~b : (b | 0x80000000u);
}
__device__ __forceinline__ float fdec(unsigned k) {
  unsigned b = (k & 0x80000000u) ? (k ^ 0x80000000u) : ~k;
  return __uint_as_float(b);
}

// ---------- WMMA fragment loads (wave32, contiguous k = 16*half + i in both A and B) ----------
// A: 16x32 bf16, row-major source. lane m=lane&15 holds row m;
// half=lane>>4 selects K = half*16 .. half*16+15
__device__ __forceinline__ v16bf load_a_frag(const __bf16* __restrict__ A,
                                             int row0, int kb, int lda, int lane) {
  int m = lane & 15, half = lane >> 4;
  const __bf16* p = A + (size_t)(row0 + m) * lda + kb + half * 16;
  v16bf a;
#pragma unroll
  for (int i = 0; i < 8; ++i) a[i] = p[i];
#pragma unroll
  for (int i = 0; i < 8; ++i) a[8 + i] = p[8 + i];
  return a;
}

// B: 32x16 bf16. lanes 0-15 hold col n=lane, K=0..15; lanes 16-31 col n=lane-16, K=16..31.
// row-major [K x N] source:
__device__ __forceinline__ v16bf load_b_frag_n(const __bf16* __restrict__ B,
                                               int col0, int kb, int ldb, int lane) {
  int nn = lane & 15, half = lane >> 4;
  const __bf16* p = B + (size_t)(kb + half * 16) * ldb + col0 + nn;
  v16bf b;
#pragma unroll
  for (int i = 0; i < 16; ++i) b[i] = p[(size_t)i * ldb];
  return b;
}
// transposed source: Bsrc is [N x K] row-major, we use Bsrc^T as the B operand
__device__ __forceinline__ v16bf load_b_frag_t(const __bf16* __restrict__ B,
                                               int col0, int kb, int ldb, int lane) {
  int nn = lane & 15, half = lane >> 4;
  const __bf16* p = B + (size_t)(col0 + nn) * ldb + kb + half * 16;
  v16bf b;
#pragma unroll
  for (int i = 0; i < 16; ++i) b[i] = p[i];
  return b;
}

// one 16x16 f32 output tile accumulated over K (multiple of 32)
template <bool BT>
__device__ __forceinline__ v8f wmma_tile(const __bf16* __restrict__ A,
                                         const __bf16* __restrict__ B,
                                         int row0, int col0, int K,
                                         int lda, int ldb, int lane) {
  v8f c = {0.f, 0.f, 0.f, 0.f, 0.f, 0.f, 0.f, 0.f};
  for (int kb = 0; kb < K; kb += 32) {
    v16bf a = load_a_frag(A, row0, kb, lda, lane);
    v16bf b = BT ? load_b_frag_t(B, col0, kb, ldb, lane)
                 : load_b_frag_n(B, col0, kb, ldb, lane);
    c = __builtin_amdgcn_wmma_f32_16x16x32_bf16(false, a, false, b, (short)0, c,
                                                false, false);
  }
  return c;
}

// 16x32 tile pair (shared A fragment, two B^T fragments, two accumulators):
// used for the N x N score passes where A reuse halves fragment traffic.
__device__ __forceinline__ void wmma_tile_pair_t(const __bf16* __restrict__ A,
                                                 const __bf16* __restrict__ B,
                                                 int row0, int col0, int K, int ld,
                                                 int lane, v8f& c0, v8f& c1) {
  c0 = (v8f){0.f, 0.f, 0.f, 0.f, 0.f, 0.f, 0.f, 0.f};
  c1 = c0;
  for (int kb = 0; kb < K; kb += 32) {
    v16bf a  = load_a_frag(A, row0, kb, ld, lane);
    v16bf b0 = load_b_frag_t(B, col0, kb, ld, lane);
    v16bf b1 = load_b_frag_t(B, col0 + 16, kb, ld, lane);
    c0 = __builtin_amdgcn_wmma_f32_16x16x32_bf16(false, a, false, b0, (short)0, c0,
                                                 false, false);
    c1 = __builtin_amdgcn_wmma_f32_16x16x32_bf16(false, a, false, b1, (short)0, c1,
                                                 false, false);
  }
}

// ---------- kernels ----------
__global__ void fill_zero_u32(unsigned* __restrict__ p, unsigned n) {
  unsigned t = blockIdx.x * blockDim.x + threadIdx.x;
  if (t < n) p[t] = 0u;
}

__global__ void cvt_f32_bf16(const float* __restrict__ s, __bf16* __restrict__ d,
                             unsigned n) {
  unsigned t = blockIdx.x * blockDim.x + threadIdx.x;
  if (t < n) d[t] = (__bf16)s[t];
}

__global__ void degree_kernel(const int* __restrict__ src, const int* __restrict__ dst,
                              float* __restrict__ cs, float* __restrict__ cd) {
  int e = blockIdx.x * blockDim.x + threadIdx.x;
  if (e < NEDGE) {
    atomicAdd(&cs[src[e]], 1.f);
    atomicAdd(&cd[dst[e]], 1.f);
  }
}

__global__ void norm_kernel(float* __restrict__ cs, float* __restrict__ cd) {
  int i = blockIdx.x * blockDim.x + threadIdx.x;
  if (i < NNODE) {
    cs[i] = rsqrtf(fmaxf(cs[i], 1.f));
    cd[i] = rsqrtf(fmaxf(cd[i], 1.f));
  }
}

// Generic WMMA GEMM: C[M x Nn] = A[M x K](bf16) @ B (bf16) with optional
// epilogue: row scale (out_norm), up to two bias vectors, relu.
template <bool BT>
__global__ void gemm_bf16_kernel(const __bf16* __restrict__ A,
                                 const __bf16* __restrict__ B,
                                 float* __restrict__ C,
                                 const float* __restrict__ bias,
                                 const float* __restrict__ bias2,
                                 const float* __restrict__ rowscale, int relu,
                                 int M, int Nn, int K, int lda, int ldb, int ldc) {
  int lane = threadIdx.x & 31;
  int wave = threadIdx.x >> 5;
  int tilesX = Nn >> 4;
  int flat = blockIdx.x * (blockDim.x >> 5) + wave;
  int total = (M >> 4) * tilesX;
  if (flat >= total) return;
  int row0 = (flat / tilesX) << 4;
  int col0 = (flat % tilesX) << 4;

  v8f c = wmma_tile<BT>(A, B, row0, col0, K, lda, ldb, lane);

  int nn = lane & 15, half = lane >> 4;
  int col = col0 + nn;
  float badd = 0.f;
  if (bias)  badd += bias[col];
  if (bias2) badd += bias2[col];
#pragma unroll
  for (int r = 0; r < 8; ++r) {
    int row = row0 + r + 8 * half;
    float v = c[r];
    if (rowscale) v *= rowscale[row];
    v += badd;
    if (relu) v = fmaxf(v, 0.f);
    C[(size_t)row * ldc + col] = v;
  }
}

// edge scatter: agg[dst] += xw[src], D=HID
__global__ void scatter_add_kernel(const float* __restrict__ x,
                                   const int* __restrict__ src,
                                   const int* __restrict__ dst,
                                   float* __restrict__ agg) {
  unsigned t = blockIdx.x * blockDim.x + threadIdx.x;
  if (t >= (unsigned)NEDGE * HID) return;
  unsigned e = t >> 7, f = t & 127u;  // HID == 128
  atomicAdd(&agg[(size_t)dst[e] * HID + f], x[(size_t)src[e] * HID + f]);
}

// h = agg * in_norm + bias (optional relu); also emit bf16 copy
__global__ void gconv_finalize_kernel(const float* __restrict__ agg,
                                      const float* __restrict__ inn,
                                      const float* __restrict__ bias, int relu,
                                      float* __restrict__ Hout,
                                      __bf16* __restrict__ Hbf) {
  int i = blockIdx.x, f = threadIdx.x;
  float v = agg[(size_t)i * HID + f] * inn[i] + bias[f];
  if (relu) v = fmaxf(v, 0.f);
  Hout[(size_t)i * HID + f] = v;
  Hbf[(size_t)i * HID + f] = (__bf16)v;
}

// pass 1 of h@h^T: per-tile-pair max -> global atomicMax (encoded)
__global__ void score_max_kernel(const __bf16* __restrict__ Hb,
                                 unsigned* __restrict__ maxkey) {
  int lane = threadIdx.x & 31;
  int wave = threadIdx.x >> 5;
  const int tilesX2 = NNODE >> 5;  // column-pair tiles
  int flat = blockIdx.x * (blockDim.x >> 5) + wave;
  if (flat >= (NNODE >> 4) * tilesX2) return;
  int row0 = (flat / tilesX2) << 4;
  int col0 = (flat % tilesX2) << 5;
  v8f c0, c1;
  wmma_tile_pair_t(Hb, Hb, row0, col0, HID, HID, lane, c0, c1);
  float m = fmaxf(c0[0], c1[0]);
#pragma unroll
  for (int r = 1; r < 8; ++r) m = fmaxf(m, fmaxf(c0[r], c1[r]));
  atomicMax(maxkey, fenc(m));
}

// straight-through Bernoulli edge predicate for off-diagonal (i,j)
__device__ __forceinline__ bool edge_pred(float s, const float* __restrict__ u,
                                          int row, int col, float sig_max) {
  int a = min(row, col), b = max(row, col);
  float P = sigmoidf_(s) / sig_max;  // ALPHA = 1
  P = fminf(fmaxf(P, EPSF), 1.f - EPSF);
  float uu = u[(size_t)a * NNODE + b];
  uu = fminf(fmaxf(uu, EPSF), 1.f - EPSF);
  // round(sigmoid(x/T)) == 1  <=>  x > 0
  return (logf(P) - log1pf(-P) + logf(uu) - log1pf(-uu)) > 0.f;
}

// pass 2: recompute score tiles, evaluate predicate, set mask bits
// (mask[i][j] = (i==j) || edge(min,max) with symmetric score)
__global__ void mask_build_kernel(const __bf16* __restrict__ Hb,
                                  const float* __restrict__ u,
                                  const unsigned* __restrict__ maxkey,
                                  unsigned* __restrict__ maskbits) {
  int lane = threadIdx.x & 31;
  int wave = threadIdx.x >> 5;
  const int tilesX2 = NNODE >> 5;
  int flat = blockIdx.x * (blockDim.x >> 5) + wave;
  if (flat >= (NNODE >> 4) * tilesX2) return;
  int row0 = (flat / tilesX2) << 4;
  int col0 = (flat % tilesX2) << 5;
  v8f c0, c1;
  wmma_tile_pair_t(Hb, Hb, row0, col0, HID, HID, lane, c0, c1);

  float sig_max = sigmoidf_(fdec(*maxkey));
  int nn = lane & 15, half = lane >> 4;
#pragma unroll
  for (int r = 0; r < 8; ++r) {
    int row = row0 + r + 8 * half;
    int colA = col0 + nn;
    int colB = col0 + 16 + nn;
    bool pA = (row == colA) || edge_pred(c0[r], u, row, colA, sig_max);
    bool pB = (row == colB) || edge_pred(c1[r], u, row, colB, sig_max);
    if (pA)
      atomicOr(&maskbits[(size_t)row * NWRD + (colA >> 5)], 1u << (colA & 31));
    if (pB)
      atomicOr(&maskbits[(size_t)row * NWRD + (colB >> 5)], 1u << (colB & 31));
  }
}

// agg[i][f] = max over set bits j of z[j][f]; blockDim.x == D.
// The mask word is uniform across the block (same row) -> force it scalar so
// bit-scan loop control runs in SALU and only v_max executes per set bit.
__global__ void masked_max_kernel(const unsigned* __restrict__ maskbits,
                                  const float* __restrict__ z,
                                  float* __restrict__ agg, int D) {
  int i = blockIdx.x, f = threadIdx.x;
  const unsigned* row = maskbits + (size_t)i * NWRD;
  float m = -INFINITY;
  for (int w = 0; w < NWRD; ++w) {
    unsigned word = (unsigned)__builtin_amdgcn_readfirstlane((int)row[w]);
    int base = w << 5;
    while (word) {
      int b = __builtin_ctz(word);
      word &= word - 1;
      m = fmaxf(m, z[(size_t)(base + b) * D + f]);
    }
  }
  agg[(size_t)i * D + f] = m;  // diag bit guarantees non-empty
}

// relu + row L2-normalize, emit bf16 for next GEMM; blockDim == HID
__global__ void relu_l2norm_kernel(const float* __restrict__ Hin,
                                   __bf16* __restrict__ Hbf) {
  __shared__ float red[HID];
  int i = blockIdx.x, f = threadIdx.x;
  float v = fmaxf(Hin[(size_t)i * HID + f], 0.f);
  red[f] = v * v;
  __syncthreads();
  for (int s = HID >> 1; s > 0; s >>= 1) {
    if (f < s) red[f] += red[f + s];
    __syncthreads();
  }
  float nrm = sqrtf(red[0]);
  Hbf[(size_t)i * HID + f] = (__bf16)(v / fmaxf(nrm, 1e-12f));
}

// ---------- host ----------
static inline int gemm_blocks(int M, int Nn) {
  int t = (M / 16) * (Nn / 16);
  return (t + 3) / 4;  // 4 waves (128 threads) per block
}

extern "C" void kernel_launch(void* const* d_in, const int* in_sizes, int n_in,
                              void* d_out, int out_size, void* d_ws, size_t ws_size,
                              hipStream_t stream) {
  (void)in_sizes; (void)n_in; (void)out_size; (void)ws_size;
  // inputs (setup_inputs order); adj (d_in[0]) is unused since ALPHA == 1
  const float* inputs = (const float*)d_in[1];
  const float* feat   = (const float*)d_in[2];
  const float* u      = (const float*)d_in[3];
  const int*   src    = (const int*)d_in[4];
  const int*   dst    = (const int*)d_in[5];
  const float* gc0_W  = (const float*)d_in[6];  const float* gc0_b = (const float*)d_in[7];
  const float* gc1_W  = (const float*)d_in[8];  const float* gc1_b = (const float*)d_in[9];
  const float* p0_W   = (const float*)d_in[10]; const float* p0_b  = (const float*)d_in[11];
  const float* l0_W   = (const float*)d_in[12]; const float* l0_b  = (const float*)d_in[13];
  const float* b0     = (const float*)d_in[14];
  const float* p1_W   = (const float*)d_in[15]; const float* p1_b  = (const float*)d_in[16];
  const float* l1_W   = (const float*)d_in[17]; const float* l1_b  = (const float*)d_in[18];
  const float* b1     = (const float*)d_in[19];
  const float* p2_W   = (const float*)d_in[20]; const float* p2_b  = (const float*)d_in[21];
  const float* l2_W   = (const float*)d_in[22]; const float* l2_b  = (const float*)d_in[23];
  const float* b2     = (const float*)d_in[24];
  float* out = (float*)d_out;

  // workspace bump allocator (256B aligned)
  char* ws = (char*)d_ws;
  size_t o = 0;
  auto alloc = [&](size_t bytes) -> char* {
    char* r = ws + o;
    o = (o + bytes + 255) & ~(size_t)255;
    return r;
  };
  float*    outn     = (float*)alloc((size_t)NNODE * 4);
  float*    inn      = (float*)alloc((size_t)NNODE * 4);
  unsigned* maxkey   = (unsigned*)alloc(256);
  unsigned* maskbits = (unsigned*)alloc((size_t)NNODE * NWRD * 4);
  size_t zero_prefix = o;  // everything above must start at zero
  __bf16* in_bf   = (__bf16*)alloc((size_t)NNODE * FIN * 2);
  __bf16* feat_bf = (__bf16*)alloc((size_t)NNODE * FIN * 2);
  __bf16* w_gc0 = (__bf16*)alloc((size_t)FIN * HID * 2);
  __bf16* w_gc1 = (__bf16*)alloc((size_t)HID * HID * 2);
  __bf16* w_p0  = (__bf16*)alloc((size_t)FIN * FIN * 2);
  __bf16* w_l0  = (__bf16*)alloc((size_t)FIN * HID * 2);
  __bf16* w_p1  = (__bf16*)alloc((size_t)HID * HID * 2);
  __bf16* w_l1  = (__bf16*)alloc((size_t)HID * HID * 2);
  __bf16* w_p2  = (__bf16*)alloc((size_t)HID * HID * 2);
  __bf16* w_l2  = (__bf16*)alloc((size_t)HID * NCLS * 2);
  float*  xw    = (float*)alloc((size_t)NNODE * HID * 4);
  float*  agg   = (float*)alloc((size_t)NNODE * HID * 4);
  float*  h1    = (float*)alloc((size_t)NNODE * HID * 4);
  __bf16* h1b   = (__bf16*)alloc((size_t)NNODE * HID * 2);
  float*  h2    = (float*)alloc((size_t)NNODE * HID * 4);
  __bf16* h2b   = (__bf16*)alloc((size_t)NNODE * HID * 2);
  float*  z     = (float*)alloc((size_t)NNODE * FIN * 4);
  float*  aggs  = (float*)alloc((size_t)NNODE * FIN * 4);
  __bf16* aggsb = (__bf16*)alloc((size_t)NNODE * FIN * 2);
  float*  hs    = (float*)alloc((size_t)NNODE * HID * 4);
  __bf16* hsb   = (__bf16*)alloc((size_t)NNODE * HID * 2);

  auto zero = [&](void* p, size_t nwords) {
    fill_zero_u32<<<(unsigned)((nwords + 255) / 256), 256, 0, stream>>>(
        (unsigned*)p, (unsigned)nwords);
  };
  auto cvt = [&](const float* s, __bf16* d, size_t n) {
    cvt_f32_bf16<<<(unsigned)((n + 255) / 256), 256, 0, stream>>>(s, d, (unsigned)n);
  };

  // --- init + conversions ---
  zero(ws, zero_prefix / 4);  // outn, inn, maxkey, maskbits
  cvt(inputs, in_bf, (size_t)NNODE * FIN);
  cvt(feat, feat_bf, (size_t)NNODE * FIN);
  cvt(gc0_W, w_gc0, (size_t)FIN * HID);
  cvt(gc1_W, w_gc1, (size_t)HID * HID);
  cvt(p0_W, w_p0, (size_t)FIN * FIN);
  cvt(l0_W, w_l0, (size_t)FIN * HID);
  cvt(p1_W, w_p1, (size_t)HID * HID);
  cvt(l1_W, w_l1, (size_t)HID * HID);
  cvt(p2_W, w_p2, (size_t)HID * HID);
  cvt(l2_W, w_l2, (size_t)HID * NCLS);

  // --- degree norms ---
  degree_kernel<<<NEDGE / 256, 256, 0, stream>>>(src, dst, outn, inn);
  norm_kernel<<<NNODE / 256, 256, 0, stream>>>(outn, inn);

  // --- GAE GraphConv 0: relu(D_in^-1/2 A D_out^-1/2 X W0 + b0) ---
  gemm_bf16_kernel<false><<<gemm_blocks(NNODE, HID), 128, 0, stream>>>(
      in_bf, w_gc0, xw, nullptr, nullptr, outn, 0, NNODE, HID, FIN, FIN, HID, HID);
  zero(agg, (size_t)NNODE * HID);
  scatter_add_kernel<<<(NEDGE * HID) / 256, 256, 0, stream>>>(xw, src, dst, agg);
  gconv_finalize_kernel<<<NNODE, HID, 0, stream>>>(agg, inn, gc0_b, 1, h1, h1b);

  // --- GAE GraphConv 1 (no relu) ---
  gemm_bf16_kernel<false><<<gemm_blocks(NNODE, HID), 128, 0, stream>>>(
      h1b, w_gc1, xw, nullptr, nullptr, outn, 0, NNODE, HID, HID, HID, HID, HID);
  zero(agg, (size_t)NNODE * HID);
  scatter_add_kernel<<<(NEDGE * HID) / 256, 256, 0, stream>>>(xw, src, dst, agg);
  gconv_finalize_kernel<<<NNODE, HID, 0, stream>>>(agg, inn, gc1_b, 0, h2, h2b);

  // --- decode + sample + mask, fused: two WMMA passes over h@h^T ---
  int ptiles = (NNODE / 16) * (NNODE / 32);  // 16x32 tile pairs
  score_max_kernel<<<(ptiles + 3) / 4, 128, 0, stream>>>(h2b, maxkey);
  mask_build_kernel<<<(ptiles + 3) / 4, 128, 0, stream>>>(h2b, u, maxkey, maskbits);

  // --- GraphSAGE layer 0 (FIN -> HID) ---
  gemm_bf16_kernel<false><<<gemm_blocks(NNODE, FIN), 128, 0, stream>>>(
      feat_bf, w_p0, z, p0_b, nullptr, nullptr, 1, NNODE, FIN, FIN, FIN, FIN, FIN);
  masked_max_kernel<<<NNODE, FIN, 0, stream>>>(maskbits, z, aggs, FIN);
  cvt(aggs, aggsb, (size_t)NNODE * FIN);
  gemm_bf16_kernel<false><<<gemm_blocks(NNODE, HID), 128, 0, stream>>>(
      aggsb, w_l0, hs, l0_b, b0, nullptr, 0, NNODE, HID, FIN, FIN, HID, HID);
  relu_l2norm_kernel<<<NNODE, HID, 0, stream>>>(hs, hsb);

  // --- GraphSAGE layer 1 (HID -> HID) ---
  gemm_bf16_kernel<false><<<gemm_blocks(NNODE, HID), 128, 0, stream>>>(
      hsb, w_p1, z, p1_b, nullptr, nullptr, 1, NNODE, HID, HID, HID, HID, HID);
  masked_max_kernel<<<NNODE, HID, 0, stream>>>(maskbits, z, aggs, HID);
  cvt(aggs, aggsb, (size_t)NNODE * HID);
  gemm_bf16_kernel<false><<<gemm_blocks(NNODE, HID), 128, 0, stream>>>(
      aggsb, w_l1, hs, l1_b, b1, nullptr, 0, NNODE, HID, HID, HID, HID, HID);
  relu_l2norm_kernel<<<NNODE, HID, 0, stream>>>(hs, hsb);

  // --- GraphSAGE layer 2 (HID -> NCLS), writes d_out ---
  gemm_bf16_kernel<false><<<gemm_blocks(NNODE, HID), 128, 0, stream>>>(
      hsb, w_p2, z, p2_b, nullptr, nullptr, 1, NNODE, HID, HID, HID, HID, HID);
  masked_max_kernel<<<NNODE, HID, 0, stream>>>(maskbits, z, aggs, HID);
  cvt(aggs, aggsb, (size_t)NNODE * HID);
  gemm_bf16_kernel<false><<<gemm_blocks(NNODE, NCLS), 128, 0, stream>>>(
      aggsb, w_l2, out, l2_b, b2, nullptr, 0, NNODE, NCLS, HID, HID, NCLS, NCLS);
}
